// SelfAttention_49727131353915
// MI455X (gfx1250) — compile-verified, matrix operand order checked by reading, not run on hardware
//
#include <hip/hip_runtime.h>
#include <hip/hip_bf16.h>

// ---------------------------------------------------------------------------
// MSA soft-tied attention with pair bias for gfx1250 (MI455X), wave32 + WMMA.
// R=128 rows, N=512, DIM=128, H=8, DH=32, INNER=256, DP=128, DHW=16.
// ---------------------------------------------------------------------------

#define R_   128
#define N_   512
#define DIM_ 128
#define H_   8
#define DH_  32
#define INNER_ 256
#define DP_  128
#define DHW_ 16

typedef _Float16 half_t;
typedef __attribute__((ext_vector_type(8)))  _Float16 v8h;
typedef __attribute__((ext_vector_type(16))) _Float16 v16h;
typedef __attribute__((ext_vector_type(8)))  float    v8f;

// ---------------------------------------------------------------------------
// WMMA helpers.
// 16-bit A/B fragment (16x32 / 32x16): per CDNA5 ISA 7.12.2, lanes 0-15 hold
// K={0..7,16..23}, lanes 16-31 hold K={8..15,24..31}; row (A) / col (B) index
// = lane%16.  With K contiguous in memory, each lane loads two 16B chunks.
// ---------------------------------------------------------------------------
__device__ __forceinline__ v16h load_frag(const half_t* row_base, int lane) {
  const half_t* p = row_base + ((lane & 16) ? 8 : 0);
  v8h lo = *(const v8h*)(p);
  v8h hi = *(const v8h*)(p + 16);
  return __builtin_shufflevector(lo, hi, 0,1,2,3,4,5,6,7,8,9,10,11,12,13,14,15);
}

__device__ __forceinline__ v8f wmma_f16(v16h a, v16h b, v8f c) {
  return __builtin_amdgcn_wmma_f32_16x16x32_f16(
      /*neg_a=*/false, a, /*neg_b=*/false, b,
      /*c_mod=*/(short)0, c, /*reuse_a=*/false, /*reuse_b=*/false);
}

// ---------------------------------------------------------------------------
// Elementwise f32 -> f16 convert.
// ---------------------------------------------------------------------------
__global__ void f32_to_f16_kernel(const float* __restrict__ in,
                                  half_t* __restrict__ out, size_t n) {
  size_t i = (size_t)blockIdx.x * blockDim.x + threadIdx.x;
  if (i < n) out[i] = (half_t)in[i];
}

// Transpose+convert weight: Wt[n*K + k] = W[k*ldw + coloff + n]  (f32 -> f16)
__global__ void wtrans_kernel(const float* __restrict__ W, int ldw, int coloff,
                              half_t* __restrict__ Wt, int K, int Nc) {
  int idx = blockIdx.x * blockDim.x + threadIdx.x;
  if (idx >= Nc * K) return;
  int n = idx / K, kk = idx - n * K;
  Wt[idx] = (half_t)W[(size_t)kk * ldw + coloff + n];
}

// ---------------------------------------------------------------------------
// Pair-bias path: LayerNorm over DP=128 then (DP)x(H) projection.
// One wave32 per (i,j) site; lane owns 4 channels (contiguous 16B load).
// bias_head layout: (H, N, N) f32.
// ---------------------------------------------------------------------------
__global__ void pair_bias_kernel(const float* __restrict__ pb,
                                 const float* __restrict__ ln_g,
                                 const float* __restrict__ ln_b,
                                 const float* __restrict__ Wpair,
                                 float* __restrict__ bias_head) {
  int lane = threadIdx.x & 31;
  int wave = blockIdx.x * (blockDim.x >> 5) + (threadIdx.x >> 5);
  int i = wave >> 9, j = wave & (N_ - 1);
  const float* p = pb + ((size_t)i * N_ + j) * DP_ + lane * 4;
  float x0 = p[0], x1 = p[1], x2 = p[2], x3 = p[3];
  float s  = x0 + x1 + x2 + x3;
  float s2 = x0*x0 + x1*x1 + x2*x2 + x3*x3;
  #pragma unroll
  for (int m = 16; m > 0; m >>= 1) {
    s  += __shfl_xor(s,  m, 32);
    s2 += __shfl_xor(s2, m, 32);
  }
  float mu   = s * (1.0f / DP_);
  float var  = s2 * (1.0f / DP_) - mu * mu;
  float rstd = rsqrtf(var + 1e-5f);
  int p0 = lane * 4;
  float n0 = (x0 - mu) * rstd * ln_g[p0+0] + ln_b[p0+0];
  float n1 = (x1 - mu) * rstd * ln_g[p0+1] + ln_b[p0+1];
  float n2 = (x2 - mu) * rstd * ln_g[p0+2] + ln_b[p0+2];
  float n3 = (x3 - mu) * rstd * ln_g[p0+3] + ln_b[p0+3];
  float acc[H_];
  #pragma unroll
  for (int h = 0; h < H_; ++h) {
    acc[h] = n0 * Wpair[(p0+0)*H_ + h] + n1 * Wpair[(p0+1)*H_ + h]
           + n2 * Wpair[(p0+2)*H_ + h] + n3 * Wpair[(p0+3)*H_ + h];
    #pragma unroll
    for (int m = 16; m > 0; m >>= 1) acc[h] += __shfl_xor(acc[h], m, 32);
  }
  if (lane == 0) {
    #pragma unroll
    for (int h = 0; h < H_; ++h)
      bias_head[((size_t)h * N_ + i) * N_ + j] = acc[h];
  }
}

// ---------------------------------------------------------------------------
// Generic WMMA GEMM: C[M x Nc] = A[M x K] * Bt[Nc x K]^T (+bias).
// A, Bt row-major f16; one wave computes a 16x32 output tile (2 accumulators).
// Output either f16 (Ch) or f32 (Cf).
// ---------------------------------------------------------------------------
__global__ void gemm_wmma_kernel(const half_t* __restrict__ A, int lda,
                                 const half_t* __restrict__ Bt, int ldb,
                                 int Nc, int K,
                                 half_t* __restrict__ Ch, float* __restrict__ Cf,
                                 int ldc, const float* __restrict__ bias) {
  int lane = threadIdx.x & 31;
  int wave = blockIdx.x * (blockDim.x >> 5) + (threadIdx.x >> 5);
  int ntiles = Nc >> 5;
  int it = wave / ntiles, jt = wave - it * ntiles;
  int m0 = it * 16, n0 = jt * 32;
  int li = lane & 15;
  const half_t* arow  = A  + (size_t)(m0 + li) * lda;
  const half_t* brow0 = Bt + (size_t)(n0 + li) * ldb;
  const half_t* brow1 = brow0 + (size_t)16 * ldb;
  v8f c0 = {}, c1 = {};
  for (int kc = 0; kc < K; kc += 32) {
    v16h a  = load_frag(arow  + kc, lane);
    v16h b0 = load_frag(brow0 + kc, lane);
    v16h b1 = load_frag(brow1 + kc, lane);
    c0 = wmma_f16(a, b0, c0);
    c1 = wmma_f16(a, b1, c1);
  }
  int mb = ((lane >> 4) << 3);
  int col0 = n0 + li, col1 = col0 + 16;
  float bb0 = bias ? bias[col0] : 0.0f;
  float bb1 = bias ? bias[col1] : 0.0f;
  #pragma unroll
  for (int v = 0; v < 8; ++v) {
    size_t row = (size_t)(m0 + mb + v) * ldc;
    float x0 = c0[v] + bb0, x1 = c1[v] + bb1;
    if (Ch) { Ch[row + col0] = (half_t)x0; Ch[row + col1] = (half_t)x1; }
    else    { Cf[row + col0] = x0;         Cf[row + col1] = x1; }
  }
}

// ---------------------------------------------------------------------------
// Row-weight softmax (PositionalWiseWeight).
// One wave per (n,h): sw[n,h,r] = <q_sw[n,h,:], k_sw[r,n,h,:]> / 4, softmax_r.
// w_row layout: (N, H, R) f32.
// ---------------------------------------------------------------------------
__global__ void rowweight_kernel(const half_t* __restrict__ qsw,
                                 const half_t* __restrict__ ksw,
                                 float* __restrict__ w_row) {
  int lane = threadIdx.x & 31;
  int wave = blockIdx.x * (blockDim.x >> 5) + (threadIdx.x >> 5);
  int n = wave >> 3, h = wave & 7;
  const half_t* q = qsw + (size_t)n * DIM_ + h * DHW_;
  float qv[DHW_];
  #pragma unroll
  for (int d = 0; d < DHW_; ++d) qv[d] = (float)q[d];
  const float inv_scale = 0.25f;  // 1/sqrt(16 + 1e-8)
  float sv[4];
  #pragma unroll
  for (int t = 0; t < 4; ++t) {
    int r = lane + t * 32;
    const half_t* kp = ksw + ((size_t)r * N_ + n) * DIM_ + h * DHW_;
    float s = 0.0f;
    #pragma unroll
    for (int d = 0; d < DHW_; ++d) s += qv[d] * (float)kp[d];
    sv[t] = s * inv_scale;
  }
  float mx = fmaxf(fmaxf(sv[0], sv[1]), fmaxf(sv[2], sv[3]));
  #pragma unroll
  for (int m = 16; m > 0; m >>= 1) mx = fmaxf(mx, __shfl_xor(mx, m, 32));
  float se = 0.0f;
  #pragma unroll
  for (int t = 0; t < 4; ++t) { sv[t] = __expf(sv[t] - mx); se += sv[t]; }
  #pragma unroll
  for (int m = 16; m > 0; m >>= 1) se += __shfl_xor(se, m, 32);
  float inv = 1.0f / se;
  float* wr = w_row + ((size_t)n * H_ + h) * R_;
  #pragma unroll
  for (int t = 0; t < 4; ++t) wr[lane + t * 32] = sv[t] * inv;
}

// In-place q[(r,i), h*32+d] *= w_row[i,h,r]
__global__ void scaleq_kernel(half_t* __restrict__ q,
                              const float* __restrict__ w_row) {
  size_t idx = (size_t)blockIdx.x * blockDim.x + threadIdx.x;  // R*N*INNER
  int c = idx & (INNER_ - 1);
  size_t t = idx >> 8;
  int i = (int)(t & (N_ - 1));
  int r = (int)(t >> 9);
  int h = c >> 5;
  float w = w_row[((size_t)i * H_ + h) * R_ + r];
  q[idx] = (half_t)((float)q[idx] * w);
}

// ---------------------------------------------------------------------------
// Soft-tied attention logits:
//   dots[h,i,j] = scale * sum_{r,d} qw[(r,i),h,d] * k[(r,j),h,d] + bias_head
// Per-head GEMM with K = R*DH = 4096; one wave computes a 32x32 tile (2x2).
// ---------------------------------------------------------------------------
__global__ void dots_wmma_kernel(const half_t* __restrict__ qw,
                                 const half_t* __restrict__ kb,
                                 const float* __restrict__ bias_head,
                                 float* __restrict__ dots) {
  int lane = threadIdx.x & 31;
  int wave = blockIdx.x * (blockDim.x >> 5) + (threadIdx.x >> 5);
  int h = wave >> 8;           // 256 (= 16x16) 32x32-tiles per head
  int t = wave & 255;
  int it = t >> 4, jt = t & 15;
  int i0 = it * 32, j0 = jt * 32;
  int li = lane & 15;
  const int hoff = h * DH_;
  v8f c00 = {}, c01 = {}, c10 = {}, c11 = {};
  for (int r = 0; r < R_; ++r) {
    const half_t* qbase = qw + (size_t)r * N_ * INNER_ + hoff;
    const half_t* kbase = kb + (size_t)r * N_ * INNER_ + hoff;
    v16h a0 = load_frag(qbase + (size_t)(i0 + li)      * INNER_, lane);
    v16h a1 = load_frag(qbase + (size_t)(i0 + 16 + li) * INNER_, lane);
    v16h b0 = load_frag(kbase + (size_t)(j0 + li)      * INNER_, lane);
    v16h b1 = load_frag(kbase + (size_t)(j0 + 16 + li) * INNER_, lane);
    c00 = wmma_f16(a0, b0, c00);
    c01 = wmma_f16(a0, b1, c01);
    c10 = wmma_f16(a1, b0, c10);
    c11 = wmma_f16(a1, b1, c11);
  }
  const float scale = 0.17677669529663687f;  // DH^-0.5
  int mb = ((lane >> 4) << 3);
  #pragma unroll
  for (int v = 0; v < 8; ++v) {
    int i_lo = i0 + mb + v, i_hi = i_lo + 16;
    int j_lo = j0 + li,     j_hi = j_lo + 16;
    size_t blo = ((size_t)h * N_ + i_lo) * N_;
    size_t bhi = ((size_t)h * N_ + i_hi) * N_;
    dots[blo + j_lo] = c00[v] * scale + bias_head[blo + j_lo];
    dots[blo + j_hi] = c01[v] * scale + bias_head[blo + j_hi];
    dots[bhi + j_lo] = c10[v] * scale + bias_head[bhi + j_lo];
    dots[bhi + j_hi] = c11[v] * scale + bias_head[bhi + j_hi];
  }
}

// Softmax over j; one wave per (h,i) row of 512, output f16.
__global__ void softmax_kernel(const float* __restrict__ dots,
                               half_t* __restrict__ attn) {
  int lane = threadIdx.x & 31;
  int wave = blockIdx.x * (blockDim.x >> 5) + (threadIdx.x >> 5);  // h*N + i
  const float* row = dots + (size_t)wave * N_;
  half_t* orow = attn + (size_t)wave * N_;
  float v[16];
  float mx = -3.0e38f;
  #pragma unroll
  for (int t = 0; t < 16; ++t) { v[t] = row[lane + t * 32]; mx = fmaxf(mx, v[t]); }
  #pragma unroll
  for (int m = 16; m > 0; m >>= 1) mx = fmaxf(mx, __shfl_xor(mx, m, 32));
  float se = 0.0f;
  #pragma unroll
  for (int t = 0; t < 16; ++t) { v[t] = __expf(v[t] - mx); se += v[t]; }
  #pragma unroll
  for (int m = 16; m > 0; m >>= 1) se += __shfl_xor(se, m, 32);
  float inv = 1.0f / se;
  #pragma unroll
  for (int t = 0; t < 16; ++t) orow[lane + t * 32] = (half_t)(v[t] * inv);
}

// v[(r,j), h*32+d] -> vt[((r*H+h)*DH + d)*N + j]  (j contiguous for B frags)
__global__ void vtrans_kernel(const half_t* __restrict__ vbuf,
                              half_t* __restrict__ vt) {
  size_t idx = (size_t)blockIdx.x * blockDim.x + threadIdx.x;  // R*H*DH*N
  int j = (int)(idx & (N_ - 1));
  size_t t = idx >> 9;
  int d = (int)(t & (DH_ - 1)); t >>= 5;
  int h = (int)(t & (H_ - 1));  t >>= 3;
  int r = (int)t;
  vt[idx] = vbuf[((size_t)r * N_ + j) * INNER_ + h * DH_ + d];
}

// ---------------------------------------------------------------------------
// attn @ v per (r,h): out[(r,i), h*32+d] = sum_j attn[h,i,j] * vt[(r,h,d), j]
// One wave per (r,h,i-tile): 16x32 output (2 accumulators), K = N = 512.
// ---------------------------------------------------------------------------
__global__ void av_wmma_kernel(const half_t* __restrict__ attn,
                               const half_t* __restrict__ vt,
                               half_t* __restrict__ out_attn) {
  int lane = threadIdx.x & 31;
  int wave = blockIdx.x * (blockDim.x >> 5) + (threadIdx.x >> 5);
  int it = wave & 31;            // 32 i-tiles
  int rh = wave >> 5;            // r*H + h
  int h = rh & 7, r = rh >> 3;
  int i0 = it * 16;
  int li = lane & 15;
  const half_t* arow  = attn + ((size_t)h * N_ + (i0 + li)) * N_;
  const half_t* brow0 = vt + ((size_t)rh * DH_ + li)        * N_;
  const half_t* brow1 = vt + ((size_t)rh * DH_ + 16 + li)   * N_;
  v8f c0 = {}, c1 = {};
  for (int jc = 0; jc < N_; jc += 32) {
    v16h a  = load_frag(arow  + jc, lane);
    v16h b0 = load_frag(brow0 + jc, lane);
    v16h b1 = load_frag(brow1 + jc, lane);
    c0 = wmma_f16(a, b0, c0);
    c1 = wmma_f16(a, b1, c1);
  }
  int mb = ((lane >> 4) << 3);
  #pragma unroll
  for (int v = 0; v < 8; ++v) {
    size_t o = ((size_t)r * N_ + (i0 + mb + v)) * INNER_ + h * DH_;
    out_attn[o + li]      = (half_t)c0[v];
    out_attn[o + 16 + li] = (half_t)c1[v];
  }
}

// ---------------------------------------------------------------------------
// Host-side orchestration.
// ---------------------------------------------------------------------------
extern "C" void kernel_launch(void* const* d_in, const int* in_sizes, int n_in,
                              void* d_out, int out_size, void* d_ws, size_t ws_size,
                              hipStream_t stream) {
  const float* x        = (const float*)d_in[0];
  const float* pairb    = (const float*)d_in[1];
  const float* Wq       = (const float*)d_in[2];
  const float* Wkv      = (const float*)d_in[3];
  const float* Wout     = (const float*)d_in[4];
  const float* b_out    = (const float*)d_in[5];
  const float* ln_g     = (const float*)d_in[6];
  const float* ln_b     = (const float*)d_in[7];
  const float* Wpair    = (const float*)d_in[8];
  const float* Wq_sw    = (const float*)d_in[9];
  const float* bq_sw    = (const float*)d_in[10];
  const float* Wk_sw    = (const float*)d_in[11];
  const float* bk_sw    = (const float*)d_in[12];
  float* out = (float*)d_out;

  // ---- workspace carve (bump allocator, 256B aligned) ----
  char* ws = (char*)d_ws;
  auto carve = [&](size_t bytes) -> char* {
    char* p = ws;
    ws += (bytes + 255) & ~(size_t)255;
    return p;
  };
  const size_t RN = (size_t)R_ * N_;              // 65536
  half_t* xh    = (half_t*)carve(RN * DIM_   * 2);   // 16.8 MB
  half_t* qbuf  = (half_t*)carve(RN * INNER_ * 2);   // 33.5 MB (q, later q*w_row)
  half_t* kbuf  = (half_t*)carve(RN * INNER_ * 2);
  half_t* vbuf  = (half_t*)carve(RN * INNER_ * 2);
  half_t* vt    = (half_t*)carve(RN * INNER_ * 2);
  half_t* oattn = (half_t*)carve(RN * INNER_ * 2);
  half_t* ksw   = (half_t*)carve(RN * DIM_   * 2);
  half_t* qsw   = (half_t*)carve((size_t)N_ * DIM_ * 2);
  float*  w_row = (float*)carve((size_t)N_ * H_ * R_ * 4);   // 2 MB
  float*  bhead = (float*)carve((size_t)H_ * N_ * N_ * 4);   // 8.4 MB
  float*  dots  = (float*)carve((size_t)H_ * N_ * N_ * 4);   // 8.4 MB
  half_t* attn  = (half_t*)carve((size_t)H_ * N_ * N_ * 2);  // 4.2 MB
  half_t* WqT   = (half_t*)carve((size_t)INNER_ * DIM_ * 2);
  half_t* WkT   = (half_t*)carve((size_t)INNER_ * DIM_ * 2);
  half_t* WvT   = (half_t*)carve((size_t)INNER_ * DIM_ * 2);
  half_t* WkswT = (half_t*)carve((size_t)DIM_ * DIM_ * 2);
  half_t* WqswT = (half_t*)carve((size_t)DIM_ * DIM_ * 2);
  half_t* WoutT = (half_t*)carve((size_t)DIM_ * INNER_ * 2);

  const int BLK = 256;  // 8 wave32s per block

  // 1) convert x to f16
  {
    size_t n = RN * DIM_;
    f32_to_f16_kernel<<<(unsigned)((n + BLK - 1) / BLK), BLK, 0, stream>>>(x, xh, n);
  }
  // 2) transpose-convert weights: Wt[n*K+k] = W[k*ld + off + n]
  auto wt = [&](const float* W, int ldw, int off, half_t* Wt, int K, int Nc) {
    int n = Nc * K;
    wtrans_kernel<<<(n + BLK - 1) / BLK, BLK, 0, stream>>>(W, ldw, off, Wt, K, Nc);
  };
  wt(Wq,    INNER_,     0, WqT,   DIM_,   INNER_);
  wt(Wkv,   2 * INNER_, 0, WkT,   DIM_,   INNER_);
  wt(Wkv,   2 * INNER_, INNER_, WvT, DIM_, INNER_);
  wt(Wk_sw, DIM_,       0, WkswT, DIM_,   DIM_);
  wt(Wq_sw, DIM_,       0, WqswT, DIM_,   DIM_);
  wt(Wout,  DIM_,       0, WoutT, INNER_, DIM_);

  // 3) pair bias: LayerNorm + head projection (memory-bound, wave per site)
  pair_bias_kernel<<<(N_ * N_) / 8, BLK, 0, stream>>>(pairb, ln_g, ln_b, Wpair, bhead);

  // 4) projections via WMMA GEMM; wave = 16x32 tile
  auto gemm = [&](const half_t* A, int lda, const half_t* Bt, int ldb,
                  int M, int Nc, int K, half_t* Ch, float* Cf, int ldc,
                  const float* bias) {
    int waves = (M / 16) * (Nc / 32);
    gemm_wmma_kernel<<<waves / 8, BLK, 0, stream>>>(A, lda, Bt, ldb, Nc, K, Ch, Cf, ldc, bias);
  };
  gemm(xh, DIM_, WqT,   DIM_, (int)RN, INNER_, DIM_, qbuf, nullptr, INNER_, nullptr);
  gemm(xh, DIM_, WkT,   DIM_, (int)RN, INNER_, DIM_, kbuf, nullptr, INNER_, nullptr);
  gemm(xh, DIM_, WvT,   DIM_, (int)RN, INNER_, DIM_, vbuf, nullptr, INNER_, nullptr);
  gemm(xh, DIM_, WkswT, DIM_, (int)RN, DIM_,   DIM_, ksw,  nullptr, DIM_,   bk_sw);
  gemm(xh, DIM_, WqswT, DIM_, N_,      DIM_,   DIM_, qsw,  nullptr, DIM_,   bq_sw);  // row 0 of MSA

  // 5) learned soft row weights (softmax over r)
  rowweight_kernel<<<(N_ * H_) / 8, BLK, 0, stream>>>(qsw, ksw, w_row);

  // 6) q <- q * w_row (in place)
  {
    size_t n = RN * INNER_;
    scaleq_kernel<<<(unsigned)(n / BLK), BLK, 0, stream>>>(qbuf, w_row);
  }

  // 7) soft-tied logits: per-head GEMM over K = R*DH = 4096, + scale + bias
  dots_wmma_kernel<<<(H_ * 256) / 8, BLK, 0, stream>>>(qbuf, kbuf, bhead, dots);

  // 8) softmax over j
  softmax_kernel<<<(H_ * N_) / 8, BLK, 0, stream>>>(dots, attn);

  // 9) v transpose for B-fragment-friendly layout
  {
    size_t n = RN * INNER_;
    vtrans_kernel<<<(unsigned)(n / BLK), BLK, 0, stream>>>(vbuf, vt);
  }

  // 10) out_attn = attn @ v per (r,h)
  av_wmma_kernel<<<(R_ * H_ * (N_ / 16)) / 8, BLK, 0, stream>>>(attn, vt, oattn);

  // 11) final projection: out = out_attn @ Wout + b_out (f32 output)
  gemm(oattn, INNER_, WoutT, INNER_, (int)RN, DIM_, INNER_, nullptr, out, DIM_, b_out);

  (void)in_sizes; (void)n_in; (void)out_size; (void)ws_size;
}
